// Action_Prediction_69733089018031
// MI455X (gfx1250) — matrix operand order repaired and checked by reading, not compiled, not run on hardware
//
#include <hip/hip_runtime.h>
#include <stdint.h>

// ---------------------------------------------------------------------------
// Problem constants (match reference exactly; segments are uniform: N/B = 128)
// ---------------------------------------------------------------------------
#define NODES   524288
#define NSEG    4096
#define DIN     128
#define HID     256
#define SEGSZ   128          // rows per segment == rows per workgroup
#define THREADS 256          // 8 wave32

typedef __attribute__((ext_vector_type(16))) __bf16       v16bf;
typedef __attribute__((ext_vector_type(8)))  float        v8f;
typedef __attribute__((ext_vector_type(4)))  unsigned int u32x4;
typedef __attribute__((ext_vector_type(4)))  float        f32x4;

union AFrag { v16bf v; u32x4 u[2]; __bf16 h[16]; };

// ---------------------------------------------------------------------------
// Dynamic-LDS layout (bytes)
// ---------------------------------------------------------------------------
#define OFF_HA   0                        // 128x256 bf16 activations (ping)
#define OFF_HB   (OFF_HA + SEGSZ*HID*2)   // 128x256 bf16 activations (pong) / X f32 tile
#define OFF_W    (OFF_HB + SEGSZ*HID*2)   // 256x256 bf16 weight panel (WT)
#define OFF_WF   (OFF_W  + HID*HID*2)     // 256 f32 final weights
#define OFF_B0   (OFF_WF + HID*4)         // 256 f32 bias 0
#define OFF_B1   (OFF_B0 + HID*4)         // 256 f32 bias 1
#define OFF_B2   (OFF_B1 + HID*4)         // 256 f32 bias 2
#define OFF_E    (OFF_B2 + HID*4)         // 128 f32 exp(logit)
#define OFF_RED  (OFF_E  + SEGSZ*4)       // 128 f32 sum-reduce scratch
#define OFF_SC   (OFF_RED+ SEGSZ*4)       // 128 f32 gumbel scores
#define OFF_IDX  (OFF_SC + SEGSZ*4)       // 128 i32 argmax indices
#define SMEM_BYTES (OFF_IDX + SEGSZ*4)    // = 268288 B  (< 320 KiB WGP LDS)

// ---------------------------------------------------------------------------
// Kernel 1: convert weights f32 -> bf16, transposed: WT[n][k] = W[k][n]
// ws layout: WT0 [HID][DIN] | WT1 [HID][HID] | WT2 [HID][HID]
// ---------------------------------------------------------------------------
__global__ void cvt_weights_kernel(const float* __restrict__ W0,
                                   const float* __restrict__ W1,
                                   const float* __restrict__ W2,
                                   __bf16* __restrict__ ws) {
  int i = blockIdx.x * blockDim.x + threadIdx.x;
  __bf16* WT0 = ws;
  __bf16* WT1 = ws + HID * DIN;
  __bf16* WT2 = WT1 + HID * HID;
  if (i < HID * DIN) {
    int n = i / DIN, k = i % DIN;
    WT0[i] = (__bf16)W0[k * HID + n];
  } else if (i < HID * DIN + HID * HID) {
    int j = i - HID * DIN;
    int n = j / HID, k = j % HID;
    WT1[j] = (__bf16)W1[k * HID + n];
  } else if (i < HID * DIN + 2 * HID * HID) {
    int j = i - HID * DIN - HID * HID;
    int n = j / HID, k = j % HID;
    WT2[j] = (__bf16)W2[k * HID + n];
  }
}

// ---------------------------------------------------------------------------
// Async global -> LDS staging (CDNA5 GLOBAL_LOAD_ASYNC_TO_LDS_B128, ASYNCcnt).
// Each lane issues 16B transfers: LDS[ldsa] = MEM[saddr + voff]  (GVS mode).
// ---------------------------------------------------------------------------
__device__ __forceinline__ void stage_async(const void* __restrict__ g,
                                            unsigned lds_off, int nbytes, int tid) {
  const int n16 = nbytes >> 4;
  for (int i = tid; i < n16; i += THREADS) {
    unsigned ldsa = lds_off + ((unsigned)i << 4);
    unsigned voff = ((unsigned)i << 4);
    asm volatile("global_load_async_to_lds_b128 %0, %1, %2"
                 :: "v"(ldsa), "v"(voff), "s"(g)
                 : "memory");
  }
}
__device__ __forceinline__ void wait_async() {
  asm volatile("s_wait_asynccnt 0" ::: "memory");
}

// ---------------------------------------------------------------------------
// Dual-M-tile GEMM strip: this wave computes M-tiles {mt0,mt1} x 8 N-tiles.
// nt loop unrolled x2 -> 4 independent WMMA accumulator chains per iteration,
// so chain epilogues (WMMA->VALU hazard: 4 co-exec slots for bf16) overlap
// other chains' WMMAs instead of v_nops, and LDS B traffic is halved vs
// one-M-tile-per-wave.
// B frag per lane: col n = nt*16+(lane&15); K = kc*32+(lane>>4)*8+{0..7} and
// +16+{0..7} from WT[n][*] -> two ds_load_b128.
// C/D: VGPR r of lane L -> row (r + 8*(L>>4)), col (L&15).
// ---------------------------------------------------------------------------
template <int K>
__device__ __forceinline__ void wmma_strip2(const AFrag* a0, const AFrag* a1,
                                            const __bf16* sW, const float* sB,
                                            __bf16* hout, int mt0, int mt1,
                                            int ntbase, int lane) {
  const int mrow = lane & 15;
  const int lhi  = lane >> 4;
#pragma unroll 2
  for (int nt = 0; nt < 8; ++nt) {
    const int n = (ntbase + nt) * 16 + mrow;
    const __bf16* wp = sW + n * K + lhi * 8;
    v8f c0 = {};
    v8f c1 = {};
#pragma unroll
    for (int kc = 0; kc < K / 32; ++kc) {
      AFrag b;
      b.u[0] = *(const u32x4*)(wp + kc * 32);
      b.u[1] = *(const u32x4*)(wp + kc * 32 + 16);
      c0 = __builtin_amdgcn_wmma_f32_16x16x32_bf16(
          false, a0[kc].v, false, b.v, (short)0, c0, false, false);
      c1 = __builtin_amdgcn_wmma_f32_16x16x32_bf16(
          false, a1[kc].v, false, b.v, (short)0, c1, false, false);
    }
    float bn = sB[n];
#pragma unroll
    for (int r = 0; r < 8; ++r) {
      float v0 = c0[r] + bn;
      float v1 = c1[r] + bn;
      v0 = v0 > 0.f ? v0 : 0.f;
      v1 = v1 > 0.f ? v1 : 0.f;
      hout[(mt0 * 16 + lhi * 8 + r) * HID + n] = (__bf16)v0;
      hout[(mt1 * 16 + lhi * 8 + r) * HID + n] = (__bf16)v1;
    }
  }
}

// A frags from LDS bf16 activations (row-major [128][K])
template <int K>
__device__ __forceinline__ void load_a_lds(AFrag* a, const __bf16* hin,
                                           int mt, int lane) {
  const int mrow = lane & 15;
  const int lhi  = lane >> 4;
  const __bf16* arow = hin + (mt * 16 + mrow) * K;
#pragma unroll
  for (int kc = 0; kc < K / 32; ++kc) {
    const __bf16* p = arow + kc * 32 + lhi * 8;
    a[kc].u[0] = *(const u32x4*)p;         // K {0..7}  (+kbase)
    a[kc].u[1] = *(const u32x4*)(p + 16);  // K {16..23}(+kbase)
  }
}

// A frags from the LDS-staged f32 X tile (converted bf16 on the fly); K = DIN
__device__ __forceinline__ void load_a_xlds(AFrag* a, const float* __restrict__ xl,
                                            int mt, int lane) {
  const int mrow = lane & 15;
  const int lhi  = lane >> 4;
  const float* xrow = xl + (mt * 16 + mrow) * DIN;
#pragma unroll
  for (int kc = 0; kc < DIN / 32; ++kc) {
    const float* p = xrow + kc * 32 + lhi * 8;
    f32x4 f0 = *(const f32x4*)(p);
    f32x4 f1 = *(const f32x4*)(p + 4);
    f32x4 f2 = *(const f32x4*)(p + 16);
    f32x4 f3 = *(const f32x4*)(p + 20);
#pragma unroll
    for (int j = 0; j < 4; ++j) {
      a[kc].h[j]      = (__bf16)f0[j];
      a[kc].h[4 + j]  = (__bf16)f1[j];
      a[kc].h[8 + j]  = (__bf16)f2[j];
      a[kc].h[12 + j] = (__bf16)f3[j];
    }
  }
}

// ---------------------------------------------------------------------------
// Main fused kernel: one workgroup == one segment (128 rows).
//   async-stage X tile + W panels into LDS (ASYNCcnt)
//   L0: X[128x128] @ W0 -> hA     (bf16 WMMA, f32 acc, +bias, ReLU)
//   L1: hA @ W1 -> hB
//   L2: hB @ W2 -> hA
//   logits = hA @ Wf + bf ; e=exp ; seg-sum ; gumbel ; seg-argmax ; outputs
// ---------------------------------------------------------------------------
__global__ __launch_bounds__(THREADS, 1)
void mlp_sample_kernel(const float* __restrict__ X,
                       const float* __restrict__ b0,
                       const float* __restrict__ b1,
                       const float* __restrict__ b2,
                       const float* __restrict__ Wf,
                       const float* __restrict__ bfin,
                       const __bf16* __restrict__ wt,
                       float* __restrict__ out) {
  extern __shared__ __attribute__((aligned(16))) char smem[];
  __bf16* hA  = (__bf16*)(smem + OFF_HA);
  __bf16* hB  = (__bf16*)(smem + OFF_HB);
  float*  xT  = (float*)(smem + OFF_HB);   // X tile aliases hB during layer 0
  __bf16* sW  = (__bf16*)(smem + OFF_W);
  float*  sWf = (float*)(smem + OFF_WF);
  float*  sB0 = (float*)(smem + OFF_B0);
  float*  sB1 = (float*)(smem + OFF_B1);
  float*  sB2 = (float*)(smem + OFF_B2);
  float*  sE  = (float*)(smem + OFF_E);
  float*  sRd = (float*)(smem + OFF_RED);
  float*  sSc = (float*)(smem + OFF_SC);
  int*    sId = (int*)(smem + OFF_IDX);

  const int tid  = threadIdx.x;
  const int wave = tid >> 5;
  const int lane = tid & 31;
  const int blk  = blockIdx.x;
  const int rowbase = blk * SEGSZ;

  // wave -> (M-tile pair, N-half)
  const int mt0    = (wave >> 1) * 2;
  const int mt1    = mt0 + 1;
  const int ntbase = (wave & 1) * 8;

  const __bf16* WT0 = wt;
  const __bf16* WT1 = wt + HID * DIN;
  const __bf16* WT2 = WT1 + HID * HID;

  if (tid < HID) {
    sWf[tid] = Wf[tid];
    sB0[tid] = b0[tid];
    sB1[tid] = b1[tid];
    sB2[tid] = b2[tid];
  }

  // ---- stage X tile (64KB f32) + WT0 panel via async DMA -------------------
  stage_async(X + (size_t)rowbase * DIN, OFF_HB, SEGSZ * DIN * 4, tid);
  stage_async(WT0, OFF_W, HID * DIN * 2, tid);
  __builtin_prefetch(WT1 + (size_t)tid * 64, 0, 1);  // warm next panel in L2
  wait_async();
  __syncthreads();

  // ---- layer 0 -------------------------------------------------------------
  {
    AFrag a0[DIN / 32], a1[DIN / 32];
    load_a_xlds(a0, xT, mt0, lane);
    load_a_xlds(a1, xT, mt1, lane);
    wmma_strip2<DIN>(a0, a1, sW, sB0, hA, mt0, mt1, ntbase, lane);
  }
  __syncthreads();

  // ---- layer 1 -------------------------------------------------------------
  stage_async(WT1, OFF_W, HID * HID * 2, tid);
  __builtin_prefetch(WT2 + (size_t)tid * 64, 0, 1);
  wait_async();
  __syncthreads();
  {
    AFrag a0[HID / 32], a1[HID / 32];
    load_a_lds<HID>(a0, hA, mt0, lane);
    load_a_lds<HID>(a1, hA, mt1, lane);
    wmma_strip2<HID>(a0, a1, sW, sB1, hB, mt0, mt1, ntbase, lane);
  }
  __syncthreads();

  // ---- layer 2 -------------------------------------------------------------
  stage_async(WT2, OFF_W, HID * HID * 2, tid);
  wait_async();
  __syncthreads();
  {
    AFrag a0[HID / 32], a1[HID / 32];
    load_a_lds<HID>(a0, hB, mt0, lane);
    load_a_lds<HID>(a1, hB, mt1, lane);
    wmma_strip2<HID>(a0, a1, sW, sB2, hA, mt0, mt1, ntbase, lane);
  }
  __syncthreads();

  // ---- final linear + segment softmax + Gumbel-max sample ------------------
  if (tid < SEGSZ) {
    union { u32x4 u; __bf16 h[8]; } q;
    const u32x4* rp = (const u32x4*)(hA + tid * HID);
    float acc = bfin[0];
#pragma unroll 4
    for (int k8 = 0; k8 < HID / 8; ++k8) {
      q.u = rp[k8];
#pragma unroll
      for (int j = 0; j < 8; ++j) acc += (float)q.h[j] * sWf[k8 * 8 + j];
    }
    float e = __expf(acc);
    sE[tid]  = e;
    sRd[tid] = e;
    // deterministic hash-based uniform -> Gumbel noise
    unsigned gi = (unsigned)(rowbase + tid);
    unsigned x = gi * 2654435761u + 0x9E3779B9u;
    x ^= x >> 16; x *= 0x85EBCA6Bu; x ^= x >> 13; x *= 0xC2B2AE35u; x ^= x >> 16;
    float u = (float)(x >> 8) * (1.0f / 16777216.0f) + 1e-20f;
    float g = -__logf(-__logf(u));
    // argmax of (log p + g) == argmax of (logit + g): -log(segsum) is constant
    sSc[tid] = acc + g;
    sId[tid] = tid;
  }
  __syncthreads();
  for (int s = SEGSZ / 2; s > 0; s >>= 1) {
    if (tid < s) {
      sRd[tid] += sRd[tid + s];
      float s1 = sSc[tid], s2 = sSc[tid + s];
      int   i2 = sId[tid + s];
      if (s2 > s1 || (s2 == s1 && i2 > sId[tid])) { sSc[tid] = s2; sId[tid] = i2; }
    }
    __syncthreads();
  }
  if (tid == 0) {
    float sum = sRd[0];
    int   w   = sId[0];
    out[blk]            = sE[w] / sum;           // p
    out[NSEG + blk]     = (float)w;              // actions (local index)
    out[2 * NSEG + blk] = (float)(rowbase + w);  // shifted_actions (global)
  }
}

// ---------------------------------------------------------------------------
// Entry
// ---------------------------------------------------------------------------
extern "C" void kernel_launch(void* const* d_in, const int* in_sizes, int n_in,
                              void* d_out, int out_size, void* d_ws, size_t ws_size,
                              hipStream_t stream) {
  const float* X   = (const float*)d_in[0];
  const float* W0  = (const float*)d_in[1];
  const float* b0  = (const float*)d_in[2];
  const float* W1  = (const float*)d_in[3];
  const float* b1  = (const float*)d_in[4];
  const float* W2  = (const float*)d_in[5];
  const float* b2  = (const float*)d_in[6];
  const float* Wf  = (const float*)d_in[7];
  const float* bf  = (const float*)d_in[8];
  // d_in[9] = batch: unused — segments are uniform 128-row contiguous blocks.
  (void)in_sizes; (void)n_in; (void)out_size; (void)ws_size;

  __bf16* wt  = (__bf16*)d_ws;       // needs HID*DIN + 2*HID*HID bf16 = 320 KiB
  float*  out = (float*)d_out;

  const int cvt_total = HID * DIN + 2 * HID * HID;
  hipLaunchKernelGGL(cvt_weights_kernel,
                     dim3((cvt_total + THREADS - 1) / THREADS), dim3(THREADS),
                     0, stream, W0, W1, W2, wt);

  hipLaunchKernelGGL(mlp_sample_kernel, dim3(NSEG), dim3(THREADS),
                     SMEM_BYTES, stream, X, b0, b1, b2, Wf, bf, wt, out);
}
